// DownConvDecimate5x_15298673508494
// MI455X (gfx1250) — matrix operand order repaired and checked by reading, not compiled, not run on hardware
//
#include <hip/hip_runtime.h>

// ---------------------------------------------------------------------------
// DownConvDecimate5x on MI455X (gfx1250, wave32, WMMA).
//
// Stages 1-4 (64ch -> 64ch, K=15, stride2) are GEMM-ified:
//   D[t=16, oc=64] = A[t=16, kk=960] x B[kk=960, oc=64],  kk = k*64 + ic
// computed with v_wmma_f32_16x16x32_bf16. Activations bf16, f32 accumulate.
// A fragments come from a bank-padded LDS [time][ic] tile (stride 72 elems ->
// conflict-free ds_load_b128); B fragments are pre-packed so one lane
// fragment = 32 contiguous bytes (clause-paired global_load_b128 from L2).
// Each wave computes two 16(t) x 64(oc) tiles so every B fragment feeds 2
// WMMAs (halves L2 weight traffic).
// ---------------------------------------------------------------------------

typedef __attribute__((ext_vector_type(16))) __bf16 v16bf;
typedef __attribute__((ext_vector_type(8)))  float  v8f;

#define LEAK 0.2f

// ---------------- workspace layout (bytes) ---------------------------------
#define OFF_A   ((size_t)0)                  // 98,304,000 (s0; reused s2,s3,mix,dt)
#define OFF_B   ((size_t)98304000)           // 49,152,000 (s1; reused s4)
#define OFF_WP  ((size_t)147456000)          //    491,520 packed WMMA weights
#define OFF_S3  (OFF_A + (size_t)25165824)   // past s2's 24,576,000 B
#define OFF_DT  (OFF_A + (size_t)33554432)   // past mixbuf's 30,720,000 B

#define SSTR  72      // padded LDS row stride (elements); 72*s keeps 16B align,
                      // bank step per m-lane = 8 -> conflict-free b128 reads
#define SROWS 280     // covers sOff(<=7) + max s (268)

static __device__ __forceinline__ float lrelu(float v) {
    return v >= 0.f ? v : LEAK * v;
}

// ---------------------------------------------------------------------------
// Pack conv_w1..4 (64,64,15) f32 -> WMMA-B fragment order, bf16.
//   K32 = 8*(L>=16)+(e&7)+16*(e>>3) ; k = kb>>1 ; ic = 32*(kb&1)+K32 ; oc = 16n+(L&15)
// ---------------------------------------------------------------------------
__global__ __launch_bounds__(256) void pack_wmma_weights(
    const float* __restrict__ w1, const float* __restrict__ w2,
    const float* __restrict__ w3, const float* __restrict__ w4,
    __bf16* __restrict__ wp)
{
    int idx = blockIdx.x * 256 + threadIdx.x;
    if (idx >= 4 * 61440) return;
    int s = idx / 61440;
    int r = idx % 61440;
    const float* w = (s == 0) ? w1 : (s == 1) ? w2 : (s == 2) ? w3 : w4;
    int e    = r & 15;
    int lane = (r >> 4) & 31;
    int kb   = (r >> 9) % 30;
    int n    = r / 15360;
    int K32  = ((lane >> 4) << 3) + (e & 7) + ((e >> 3) << 4); // 0..31
    int k    = kb >> 1;                           // 0..14
    int ic   = ((kb & 1) << 5) + K32;             // 0..63
    int oc   = n * 16 + (lane & 15);              // 0..63
    wp[idx] = (__bf16)w[(oc * 64 + ic) * 15 + k];
}

// ---------------------------------------------------------------------------
// Stage 0: 1ch -> 64ch, stride 2, K=15, LeakyReLU (VALU, ~1.5 GFLOP).
// ---------------------------------------------------------------------------
__global__ __launch_bounds__(256) void conv1d_first(
    const float* __restrict__ x, const float* __restrict__ w0,
    const float* __restrict__ b0, __bf16* __restrict__ out,
    int T_in, int T_out)
{
    __shared__ float sw[960];
    __shared__ float sb[64];
    __shared__ float sx[528];
    int tid = threadIdx.x, row = blockIdx.y;
    int T0 = blockIdx.x * 256;
    const float* xr = x + (size_t)row * T_in;
    for (int i = tid; i < 960; i += 256) sw[i] = w0[i];
    if (tid < 64) sb[tid] = b0[tid];
    for (int i = tid; i < 525; i += 256) {
        int g = 2 * T0 - 7 + i;
        sx[i] = ((unsigned)g < (unsigned)T_in) ? xr[g] : 0.f;
    }
    __syncthreads();

    int t = T0 + tid;
    bool valid = t < T_out;
    float xs[15];
#pragma unroll
    for (int k = 0; k < 15; ++k) xs[k] = sx[2 * tid + k];

    __bf16* orow = out + (size_t)row * 64 * T_out;
    for (int oc = 0; oc < 64; ++oc) {
        float a = sb[oc];
#pragma unroll
        for (int k = 0; k < 15; ++k) a = fmaf(sw[oc * 15 + k], xs[k], a);
        if (valid) orow[(size_t)oc * T_out + t] = (__bf16)lrelu(a);
    }
}

// ---------------------------------------------------------------------------
// Stages 1-4 via WMMA. Block = 128 threads = 4 waves; wave w computes tiles
// 2w, 2w+1 (each 16 t x 64 oc). LDS input tile staged with aligned 16B
// global loads (window start aligned down to 8 samples).
// ---------------------------------------------------------------------------
__global__ __launch_bounds__(128) void conv1d_s2_wmma(
    const __bf16* __restrict__ in, const __bf16* __restrict__ wp,
    const float* __restrict__ bias, __bf16* __restrict__ out,
    int T_in, int T_out)
{
    __shared__ __bf16 sIn[SROWS * SSTR];        // 40,320 B

    int tid = threadIdx.x;
    int row = blockIdx.y;
    int T0  = blockIdx.x * 128;                 // 8 tiles of 16 outputs
    const __bf16* inRow = in + (size_t)row * 64 * T_in;

    // ---- stage input chunk: LDS[s][ic] = in[ic][g0 + s], zero-padded ------
    int gbase = 2 * T0 - 7;
    int g0    = gbase & ~7;                     // floor-align to 8 samples
    int sOff  = gbase - g0;                     // 0..7
    for (int c = tid; c < 64 * 35; c += 128) {
        int ic  = c / 35;
        int j   = c % 35;
        int gs  = g0 + j * 8;
        int sr  = j * 8;
        int gb  = ic * T_in;
        if (gs >= 0 && gs + 8 <= T_in) {        // aligned 16B fast path
            union { uint4 q; __bf16 h[8]; } u;
            u.q = *(const uint4*)&inRow[gb + gs];
#pragma unroll
            for (int r = 0; r < 8; ++r) sIn[(sr + r) * SSTR + ic] = u.h[r];
        } else {
#pragma unroll
            for (int r = 0; r < 8; ++r) {
                int g = gs + r;
                sIn[(sr + r) * SSTR + ic] =
                    ((unsigned)g < (unsigned)T_in) ? inRow[gb + g] : (__bf16)0.f;
            }
        }
    }
    __syncthreads();

    int wave = tid >> 5, lane = tid & 31;
    int m = lane & 15, half = lane >> 4;
    int tloc0 = wave * 32;                      // first of the wave's 2 tiles

    const v16bf* wpv = (const v16bf*)wp;        // fragment: (n*30+kb)*32 + lane

    v8f a00 = {}, a01 = {}, a02 = {}, a03 = {}; // tile0, n=0..3
    v8f a10 = {}, a11 = {}, a12 = {}, a13 = {}; // tile1, n=0..3

#pragma unroll 2
    for (int kb = 0; kb < 30; ++kb) {
        int k   = kb >> 1;
        int icb = (kb & 1) << 5;
        int s0  = sOff + 2 * (tloc0 + m) + k;
        int p0  = s0 * SSTR + icb + 8 * half;   // 16B aligned
        int p1  = p0 + 32 * SSTR;               // tile1: +16 outputs -> +32 rows
        union { uint4 q[2]; v16bf v; } A0, A1;
        A0.q[0] = *(const uint4*)&sIn[p0];
        A0.q[1] = *(const uint4*)&sIn[p0 + 16];
        A1.q[0] = *(const uint4*)&sIn[p1];
        A1.q[1] = *(const uint4*)&sIn[p1 + 16];

        int fb = kb * 32 + lane;
        v16bf b0 = wpv[fb          ];
        v16bf b1 = wpv[fb + 30 * 32];
        v16bf b2 = wpv[fb + 60 * 32];
        v16bf b3 = wpv[fb + 90 * 32];

        a00 = __builtin_amdgcn_wmma_f32_16x16x32_bf16(false, A0.v, false, b0, (short)0, a00, false, false);
        a10 = __builtin_amdgcn_wmma_f32_16x16x32_bf16(false, A1.v, false, b0, (short)0, a10, false, false);
        a01 = __builtin_amdgcn_wmma_f32_16x16x32_bf16(false, A0.v, false, b1, (short)0, a01, false, false);
        a11 = __builtin_amdgcn_wmma_f32_16x16x32_bf16(false, A1.v, false, b1, (short)0, a11, false, false);
        a02 = __builtin_amdgcn_wmma_f32_16x16x32_bf16(false, A0.v, false, b2, (short)0, a02, false, false);
        a12 = __builtin_amdgcn_wmma_f32_16x16x32_bf16(false, A1.v, false, b2, (short)0, a12, false, false);
        a03 = __builtin_amdgcn_wmma_f32_16x16x32_bf16(false, A0.v, false, b3, (short)0, a03, false, false);
        a13 = __builtin_amdgcn_wmma_f32_16x16x32_bf16(false, A1.v, false, b3, (short)0, a13, false, false);
    }

    // ---- epilogue: lane holds 8 consecutive t for oc = 16n + (lane&15) ----
    __bf16* orow = out + (size_t)row * 64 * T_out;
    bool t8 = (T_out & 7) == 0;                 // uniform
    auto emit = [&](v8f acc, int n, int tloc) {
        int oc = n * 16 + m;
        float bv = bias[oc];
        int t0s = T0 + tloc + 8 * half;
        union { __bf16 h[8]; uint2 d[2]; uint4 q; } o;
#pragma unroll
        for (int i = 0; i < 8; ++i) o.h[i] = (__bf16)lrelu(acc[i] + bv);
        __bf16* p = orow + (size_t)oc * T_out + t0s;
        if (t0s + 8 <= T_out) {
            if (t8) { *(uint4*)p = o.q; }       // 16B aligned when T_out%8==0
            else    { *(uint2*)p = o.d[0]; *(uint2*)(p + 4) = o.d[1]; }
        } else {
#pragma unroll
            for (int i = 0; i < 8; ++i)
                if (t0s + i < T_out) p[i] = o.h[i];
        }
    };
    emit(a00, 0, tloc0);      emit(a01, 1, tloc0);
    emit(a02, 2, tloc0);      emit(a03, 3, tloc0);
    emit(a10, 0, tloc0 + 16); emit(a11, 1, tloc0 + 16);
    emit(a12, 2, tloc0 + 16); emit(a13, 3, tloc0 + 16);
}

// ---------------------------------------------------------------------------
// mix: (8,4,64,1500) -> (8,10,64,1500), 3x3 pad 1, LeakyReLU. f32 out.
// ---------------------------------------------------------------------------
__global__ __launch_bounds__(256) void mix_conv(
    const __bf16* __restrict__ s4, const float* __restrict__ mw,
    const float* __restrict__ mb, float* __restrict__ outb)
{
    __shared__ float sw[360];
    int tid = threadIdx.x;
    for (int i = tid; i < 360; i += 256) sw[i] = mw[i];
    __syncthreads();

    int idx = blockIdx.x * 256 + tid;
    if (idx >= 8 * 10 * 64 * 1500) return;
    int j  = idx % 1500;
    int i  = (idx / 1500) % 64;
    int hc = (idx / (1500 * 64)) % 10;
    int b  = idx / (1500 * 64 * 10);

    float acc = mb[hc];
#pragma unroll
    for (int c = 0; c < 4; ++c) {
        const __bf16* hb = s4 + ((size_t)(b * 4 + c)) * 64 * 1500;
#pragma unroll
        for (int di = 0; di < 3; ++di) {
            int ii = i + di - 1;
            if ((unsigned)ii >= 64u) continue;
#pragma unroll
            for (int dj = 0; dj < 3; ++dj) {
                int jj = j + dj - 1;
                if ((unsigned)jj >= 1500u) continue;
                acc = fmaf(sw[((hc * 4 + c) * 3 + di) * 3 + dj],
                           (float)hb[ii * 1500 + jj], acc);
            }
        }
    }
    outb[idx] = lrelu(acc);
}

// ---------------------------------------------------------------------------
// dt: (8,10,64,1500) -> (8,10,64,100), 3x15 pad (1,0), stride (1,15), lrelu.
// ---------------------------------------------------------------------------
__global__ __launch_bounds__(256) void dt_conv(
    const float* __restrict__ mixb, const float* __restrict__ dw,
    const float* __restrict__ db, float* __restrict__ outb)
{
    __shared__ float sdw[4500];
    int tid = threadIdx.x;
    for (int i = tid; i < 4500; i += 256) sdw[i] = dw[i];
    __syncthreads();

    int idx = blockIdx.x * 256 + tid;
    if (idx >= 8 * 10 * 64 * 100) return;
    int jo = idx % 100;
    int i  = (idx / 100) % 64;
    int hc = (idx / 6400) % 10;
    int b  = idx / 64000;

    float acc = db[hc];
    for (int c2 = 0; c2 < 10; ++c2) {
#pragma unroll
        for (int di = 0; di < 3; ++di) {
            int ii = i + di - 1;
            if ((unsigned)ii >= 64u) continue;
            const float* mr = mixb + (((size_t)(b * 10 + c2) * 64 + ii) * 1500
                                      + jo * 15);
            const float* wr = &sdw[((hc * 10 + c2) * 3 + di) * 15];
#pragma unroll
            for (int w = 0; w < 15; ++w) acc = fmaf(wr[w], mr[w], acc);
        }
    }
    outb[idx] = lrelu(acc);
}

// ---------------------------------------------------------------------------
// p (1x1) + transpose: out (8,10,100,64) f32.
// ---------------------------------------------------------------------------
__global__ __launch_bounds__(256) void p_conv_t(
    const float* __restrict__ dtb, const float* __restrict__ pw,
    const float* __restrict__ pb, float* __restrict__ out)
{
    int idx = blockIdx.x * 256 + threadIdx.x;
    if (idx >= 8 * 10 * 100 * 64) return;
    int i  = idx & 63;
    int jo = (idx >> 6) % 100;
    int p  = (idx / 6400) % 10;
    int b  = idx / 64000;

    float acc = pb[p];
#pragma unroll
    for (int hc = 0; hc < 10; ++hc)
        acc = fmaf(pw[p * 10 + hc],
                   dtb[((size_t)(b * 10 + hc) * 64 + i) * 100 + jo], acc);
    out[idx] = acc;
}

// ---------------------------------------------------------------------------
extern "C" void kernel_launch(void* const* d_in, const int* in_sizes, int n_in,
                              void* d_out, int out_size, void* d_ws, size_t ws_size,
                              hipStream_t stream)
{
    (void)in_sizes; (void)n_in; (void)out_size; (void)ws_size;
    const float* x    = (const float*)d_in[0];
    const float* cw0  = (const float*)d_in[1];
    const float* cb0  = (const float*)d_in[2];
    const float* cw1  = (const float*)d_in[3];
    const float* cb1  = (const float*)d_in[4];
    const float* cw2  = (const float*)d_in[5];
    const float* cb2  = (const float*)d_in[6];
    const float* cw3  = (const float*)d_in[7];
    const float* cb3  = (const float*)d_in[8];
    const float* cw4  = (const float*)d_in[9];
    const float* cb4  = (const float*)d_in[10];
    const float* mw   = (const float*)d_in[11];
    const float* mb   = (const float*)d_in[12];
    const float* dw   = (const float*)d_in[13];
    const float* db   = (const float*)d_in[14];
    const float* pw   = (const float*)d_in[15];
    const float* pb   = (const float*)d_in[16];

    char* ws = (char*)d_ws;
    __bf16* bufA = (__bf16*)(ws + OFF_A);
    __bf16* bufB = (__bf16*)(ws + OFF_B);
    __bf16* wp   = (__bf16*)(ws + OFF_WP);
    __bf16* s3o  = (__bf16*)(ws + OFF_S3);
    float*  mixb = (float*)(ws + OFF_A);
    float*  dtb  = (float*)(ws + OFF_DT);
    float*  out  = (float*)d_out;

    // 0) pack stage-1..4 weights into WMMA-B fragment order
    pack_wmma_weights<<<960, 256, 0, stream>>>(cw1, cw2, cw3, cw4, wp);

    // 1) stage 0 (1ch -> 64ch): x -> bufA   [32][64][24000] bf16
    conv1d_first<<<dim3(94, 32), 256, 0, stream>>>(x, cw0, cb0, bufA, 48000, 24000);

    // 2) WMMA stages (block = 128 threads = 4 waves, 2 t-tiles per wave)
    conv1d_s2_wmma<<<dim3(94, 32), 128, 0, stream>>>(bufA, wp + 0 * 61440, cb1,
                                                     bufB, 24000, 12000);
    conv1d_s2_wmma<<<dim3(47, 32), 128, 0, stream>>>(bufB, wp + 1 * 61440, cb2,
                                                     bufA, 12000, 6000);
    conv1d_s2_wmma<<<dim3(24, 32), 128, 0, stream>>>(bufA, wp + 2 * 61440, cb3,
                                                     s3o, 6000, 3000);
    conv1d_s2_wmma<<<dim3(12, 32), 128, 0, stream>>>(s3o, wp + 3 * 61440, cb4,
                                                     bufB, 3000, 1500);

    // 3) 2D tail
    mix_conv<<<30000, 256, 0, stream>>>(bufB, mw, mb, mixb);
    dt_conv<<<2000, 256, 0, stream>>>(mixb, dw, db, dtb);
    p_conv_t<<<2000, 256, 0, stream>>>(dtb, pw, pb, out);
}
